// GaussianAttention_40759239639763
// MI455X (gfx1250) — matrix operand order repaired and checked by reading, not compiled, not run on hardware
//
#include <hip/hip_runtime.h>
#include <hip/hip_bf16.h>

// Dims from the reference
#define TSN   256
#define CTS   512
#define BATCH 64
#define HID   512
#define KMIX  10

typedef __bf16 bf16_t;
typedef bf16_t v16bf __attribute__((ext_vector_type(16)));
typedef bf16_t v8bf  __attribute__((ext_vector_type(8)));
typedef float  v8f   __attribute__((ext_vector_type(8)));

// Pre-swizzled weight fragments: [mat(2)][gate(3)][ctile(32)][kk(16)][lane(32)][e(16)]
#define WFRAG_ELEMS ((size_t)2 * 3 * 32 * 16 * 32 * 16)   // 1,572,864 bf16 = 3 MB

__device__ __forceinline__ float sigmoid_f(float x) { return 1.0f / (1.0f + __expf(-x)); }

// ---------------------------------------------------------------------------
// One-time prep: convert GRU weights f32 -> bf16 in WMMA B-fragment order.
// B-fragment order used here (matches load_a_frag, not AMD's table order): lane -> column n = lane%16,
// K = (lane/16)*16 + e  => each lane-slot holds 16 consecutive k of one row.
// ---------------------------------------------------------------------------
__global__ __launch_bounds__(256)
void prep_weights(const float* __restrict__ Wih, const float* __restrict__ Whh,
                  bf16_t* __restrict__ swW)
{
  const int tau = blockIdx.x * 256 + threadIdx.x;   // 0 .. 98303 lane-slots
  const int lane  = tau & 31;
  const int kkI   = (tau >> 5) & 15;
  const int ctile = (tau >> 9) & 31;
  const int mg    = tau >> 14;                      // mat*3 + gate, 0..5
  const int gate  = mg % 3;
  const float* W  = (mg >= 3) ? Whh : Wih;

  const int row = gate * HID + ctile * 16 + (lane & 15);
  const int col = kkI * 32 + (lane >> 4) * 16;

  const float4* p = (const float4*)(W + (size_t)row * HID + col);
  float4 q0 = p[0], q1 = p[1], q2 = p[2], q3 = p[3];
  v16bf r;
  r[0]  = (bf16_t)q0.x; r[1]  = (bf16_t)q0.y; r[2]  = (bf16_t)q0.z; r[3]  = (bf16_t)q0.w;
  r[4]  = (bf16_t)q1.x; r[5]  = (bf16_t)q1.y; r[6]  = (bf16_t)q1.z; r[7]  = (bf16_t)q1.w;
  r[8]  = (bf16_t)q2.x; r[9]  = (bf16_t)q2.y; r[10] = (bf16_t)q2.z; r[11] = (bf16_t)q2.w;
  r[12] = (bf16_t)q3.x; r[13] = (bf16_t)q3.y; r[14] = (bf16_t)q3.z; r[15] = (bf16_t)q3.w;
  *(v16bf*)(swW + (size_t)tau * 16) = r;
}

// One-time prep: bf16 copy of the initial hidden state.
__global__ __launch_bounds__(256)
void prep_h0(const float* __restrict__ g, bf16_t* __restrict__ hb)
{
  const int i = blockIdx.x * 256 + threadIdx.x;     // BATCH*HID threads
  hb[i] = (bf16_t)g[i];
}

// ---------------------------------------------------------------------------
// Attention step. Grid: BATCH blocks x 256 threads. c_inp is L2-resident.
// Also emits a bf16 copy of w_t for the WMMA A-path (if w_bf != nullptr).
// ---------------------------------------------------------------------------
__global__ __launch_bounds__(256)
void attn_step(const float* __restrict__ x,      // [B,H] = inp[t]
               const float* __restrict__ Wa, const float* __restrict__ ba,
               const float* __restrict__ Wb, const float* __restrict__ bb,
               const float* __restrict__ Wk, const float* __restrict__ bk,
               const float* __restrict__ kprev,  // [B,K]
               const float* __restrict__ c_inp,  // [CTS,B,H]
               float* __restrict__ k_out,        // [B,K]  (att_k[t])
               float* __restrict__ w_out,        // [B,H]  (att_w[t])
               bf16_t* __restrict__ w_bf)        // [B,H]  bf16 mirror (or null)
{
  __shared__ float xs[HID];
  __shared__ float phi[CTS];
  __shared__ float a_s[KMIX], b_s[KMIX], k_s[KMIX];

  const int b    = blockIdx.x;
  const int tid  = threadIdx.x;
  const int wave = tid >> 5;
  const int lane = tid & 31;

  xs[tid]       = x[b * HID + tid];
  xs[tid + 256] = x[b * HID + 256 + tid];
  __syncthreads();

  for (int idx = wave; idx < 3 * KMIX; idx += 8) {
    const int g = idx / KMIX, comp = idx % KMIX;
    const float* Wrow = (g == 0 ? Wa : (g == 1 ? Wb : Wk)) + comp * HID;
    float s = 0.0f;
    for (int j = lane; j < HID; j += 32) s += xs[j] * Wrow[j];
    #pragma unroll
    for (int off = 16; off > 0; off >>= 1) s += __shfl_down(s, off, 32);
    if (lane == 0) {
      if (g == 0)      a_s[comp] = __expf(s + ba[comp]);
      else if (g == 1) b_s[comp] = __expf(s + bb[comp]);
      else {
        float kn = kprev[b * KMIX + comp] + __expf(s + bk[comp]);
        k_s[comp] = kn;
        k_out[b * KMIX + comp] = kn;   // recurrent k state lives in d_out
      }
    }
  }
  __syncthreads();

  for (int s = tid; s < CTS; s += 256) {
    const float fs = (float)s;
    float acc = 0.0f;
    #pragma unroll
    for (int kk = 0; kk < KMIX; ++kk) {
      float d = k_s[kk] - fs;
      acc += a_s[kk] * __expf(-b_s[kk] * d * d);
    }
    phi[s] = acc;
  }
  __syncthreads();

  const float* cb = c_inp + (size_t)b * HID;
  for (int h = tid; h < HID; h += 256) {
    float acc = 0.0f;
    for (int s = 0; s < CTS; s += 4) {
      acc += phi[s + 0] * cb[(size_t)(s + 0) * BATCH * HID + h];
      acc += phi[s + 1] * cb[(size_t)(s + 1) * BATCH * HID + h];
      acc += phi[s + 2] * cb[(size_t)(s + 2) * BATCH * HID + h];
      acc += phi[s + 3] * cb[(size_t)(s + 3) * BATCH * HID + h];
    }
    w_out[b * HID + h] = acc;
    if (w_bf) w_bf[b * HID + h] = (bf16_t)acc;
  }
}

// A-fragment (16x32 bf16) from a row-major bf16 matrix [.,HID]:
// lane -> row; elements 0..15: K = kk + half*16 + e.
__device__ __forceinline__ v16bf load_a_frag(const bf16_t* __restrict__ M,
                                             int row, int kk, int half)
{
  const v8bf r1 = *(const v8bf*)(M + (size_t)row * HID + kk + half * 16);
  const v8bf r2 = *(const v8bf*)(M + (size_t)row * HID + kk + half * 16 + 8);
  return __builtin_shufflevector(r1, r2, 0, 1, 2, 3, 4, 5, 6, 7,
                                         8, 9, 10, 11, 12, 13, 14, 15);
}

// ---------------------------------------------------------------------------
// GRU step (fast path): bf16 A/B operands, f32 WMMA accumulation.
// Grid (B/16, H/16), single-wave blocks; 6 WMMAs per K=32 slice, 96 total.
// B-fragments come pre-swizzled from d_ws: one 32B/lane coalesced load each.
// ---------------------------------------------------------------------------
__global__ __launch_bounds__(32)
void gru_step_bf(const bf16_t* __restrict__ w_bf,   // [B,H] bf16 att_w[t]
                 const bf16_t* __restrict__ h_bf,   // [B,H] bf16 h_{t-1}
                 const float*  __restrict__ hprev,  // [B,H] f32 h_{t-1} (for blend)
                 const bf16_t* __restrict__ swW,    // swizzled weight fragments
                 const float*  __restrict__ bih, const float* __restrict__ bhh,
                 float*  __restrict__ hout,         // [B,H] hiddens[t]
                 bf16_t* __restrict__ hout_bf)      // [B,H] bf16 h_t for t+1
{
  const int m0    = blockIdx.x * 16;
  const int ctile = blockIdx.y;
  const int c     = ctile * 16;
  const int lane  = threadIdx.x;
  const int half  = lane >> 4;
  const int l16   = lane & 15;

  // Per-(mat,gate) fragment bases for this column tile; +512 elems per kk step.
  const size_t fragStride = (size_t)32 * 16;                 // elems per (kk) group
  const size_t laneOfs    = (size_t)lane * 16;
  #define FRAG_BASE(mat, gate) \
    (swW + ((((size_t)(mat) * 3 + (gate)) * 32 + ctile) * 16) * fragStride / 16 * 16 + laneOfs)
  const bf16_t* p_ir = FRAG_BASE(0, 0);
  const bf16_t* p_iz = FRAG_BASE(0, 1);
  const bf16_t* p_in = FRAG_BASE(0, 2);
  const bf16_t* p_hr = FRAG_BASE(1, 0);
  const bf16_t* p_hz = FRAG_BASE(1, 1);
  const bf16_t* p_hn = FRAG_BASE(1, 2);
  #undef FRAG_BASE

  v8f acc_r = {}, acc_z = {}, acc_in = {}, acc_hn = {};

  #pragma unroll 4
  for (int kkI = 0; kkI < HID / 32; ++kkI) {
    const int kk = kkI * 32;
    const size_t fo = (size_t)kkI * fragStride;

    const v16bf a_w = load_a_frag(w_bf, m0 + l16, kk, half);
    const v16bf a_h = load_a_frag(h_bf, m0 + l16, kk, half);

    const v16bf b_ir = *(const v16bf*)(p_ir + fo);
    const v16bf b_hr = *(const v16bf*)(p_hr + fo);
    const v16bf b_iz = *(const v16bf*)(p_iz + fo);
    const v16bf b_hz = *(const v16bf*)(p_hz + fo);
    const v16bf b_in = *(const v16bf*)(p_in + fo);
    const v16bf b_hn = *(const v16bf*)(p_hn + fo);

    acc_r  = __builtin_amdgcn_wmma_f32_16x16x32_bf16(false, a_w, false, b_ir, (short)0, acc_r,  false, false);
    acc_r  = __builtin_amdgcn_wmma_f32_16x16x32_bf16(false, a_h, false, b_hr, (short)0, acc_r,  false, false);
    acc_z  = __builtin_amdgcn_wmma_f32_16x16x32_bf16(false, a_w, false, b_iz, (short)0, acc_z,  false, false);
    acc_z  = __builtin_amdgcn_wmma_f32_16x16x32_bf16(false, a_h, false, b_hz, (short)0, acc_z,  false, false);
    acc_in = __builtin_amdgcn_wmma_f32_16x16x32_bf16(false, a_w, false, b_in, (short)0, acc_in, false, false);
    acc_hn = __builtin_amdgcn_wmma_f32_16x16x32_bf16(false, a_h, false, b_hn, (short)0, acc_hn, false, false);
  }

  // Gate math on the C/D layout: element j -> M = m0 + j + half*8, N = c + l16.
  const int col = c + l16;
  const float br = bih[col]           + bhh[col];
  const float bz = bih[HID + col]     + bhh[HID + col];
  const float bi = bih[2 * HID + col];
  const float bh = bhh[2 * HID + col];

  #pragma unroll
  for (int j = 0; j < 8; ++j) {
    const int m = m0 + j + half * 8;
    const float r = sigmoid_f(acc_r[j] + br);
    const float z = sigmoid_f(acc_z[j] + bz);
    const float n = tanhf(acc_in[j] + bi + r * (acc_hn[j] + bh));
    const float hp = hprev[(size_t)m * HID + col];
    const float hv = (1.0f - z) * n + z * hp;
    hout[(size_t)m * HID + col]    = hv;
    hout_bf[(size_t)m * HID + col] = (bf16_t)hv;
  }
}

// ---------------------------------------------------------------------------
// Fallback GRU step (ws too small): convert f32 operands to bf16 in-loop.
// ---------------------------------------------------------------------------
__device__ __forceinline__ v16bf cvt16(float4 q0, float4 q1, float4 q2, float4 q3) {
  v16bf r;
  r[0]  = (bf16_t)q0.x; r[1]  = (bf16_t)q0.y; r[2]  = (bf16_t)q0.z; r[3]  = (bf16_t)q0.w;
  r[4]  = (bf16_t)q1.x; r[5]  = (bf16_t)q1.y; r[6]  = (bf16_t)q1.z; r[7]  = (bf16_t)q1.w;
  r[8]  = (bf16_t)q2.x; r[9]  = (bf16_t)q2.y; r[10] = (bf16_t)q2.z; r[11] = (bf16_t)q2.w;
  r[12] = (bf16_t)q3.x; r[13] = (bf16_t)q3.y; r[14] = (bf16_t)q3.z; r[15] = (bf16_t)q3.w;
  return r;
}

__device__ __forceinline__ v16bf load_b_frag_f32(const float* __restrict__ W,
                                                 int rowbase, int kofs,
                                                 int l16, int half) {
  const float4* p = (const float4*)(W + (size_t)(rowbase + l16) * HID + kofs + half * 16);
  return cvt16(p[0], p[1], p[2], p[3]);
}

__device__ __forceinline__ v16bf load_a_frag_f32(const float* __restrict__ M,
                                                 int row, int kk, int half) {
  const float* ar = M + (size_t)row * HID + kk;
  float4 f0 = *(const float4*)(ar + half * 16);
  float4 f1 = *(const float4*)(ar + half * 16 + 4);
  float4 f2 = *(const float4*)(ar + half * 16 + 8);
  float4 f3 = *(const float4*)(ar + half * 16 + 12);
  return cvt16(f0, f1, f2, f3);
}

__global__ __launch_bounds__(32)
void gru_step_f32(const float* __restrict__ w, const float* __restrict__ hprev,
                  const float* __restrict__ Wih, const float* __restrict__ bih,
                  const float* __restrict__ Whh, const float* __restrict__ bhh,
                  float* __restrict__ hout)
{
  const int m0 = blockIdx.x * 16, c = blockIdx.y * 16;
  const int lane = threadIdx.x, half = lane >> 4, l16 = lane & 15;

  v8f acc_r = {}, acc_z = {}, acc_in = {}, acc_hn = {};
  for (int kk = 0; kk < HID; kk += 32) {
    const v16bf a_w = load_a_frag_f32(w, m0 + l16, kk, half);
    const v16bf a_h = load_a_frag_f32(hprev, m0 + l16, kk, half);
    const v16bf b_ir = load_b_frag_f32(Wih, c,           kk, l16, half);
    const v16bf b_hr = load_b_frag_f32(Whh, c,           kk, l16, half);
    const v16bf b_iz = load_b_frag_f32(Wih, HID + c,     kk, l16, half);
    const v16bf b_hz = load_b_frag_f32(Whh, HID + c,     kk, l16, half);
    const v16bf b_in = load_b_frag_f32(Wih, 2 * HID + c, kk, l16, half);
    const v16bf b_hn = load_b_frag_f32(Whh, 2 * HID + c, kk, l16, half);
    acc_r  = __builtin_amdgcn_wmma_f32_16x16x32_bf16(false, a_w, false, b_ir, (short)0, acc_r,  false, false);
    acc_r  = __builtin_amdgcn_wmma_f32_16x16x32_bf16(false, a_h, false, b_hr, (short)0, acc_r,  false, false);
    acc_z  = __builtin_amdgcn_wmma_f32_16x16x32_bf16(false, a_w, false, b_iz, (short)0, acc_z,  false, false);
    acc_z  = __builtin_amdgcn_wmma_f32_16x16x32_bf16(false, a_h, false, b_hz, (short)0, acc_z,  false, false);
    acc_in = __builtin_amdgcn_wmma_f32_16x16x32_bf16(false, a_w, false, b_in, (short)0, acc_in, false, false);
    acc_hn = __builtin_amdgcn_wmma_f32_16x16x32_bf16(false, a_h, false, b_hn, (short)0, acc_hn, false, false);
  }
  const int col = c + l16;
  const float br = bih[col] + bhh[col];
  const float bz = bih[HID + col] + bhh[HID + col];
  const float bi = bih[2 * HID + col];
  const float bh = bhh[2 * HID + col];
  #pragma unroll
  for (int j = 0; j < 8; ++j) {
    const int m = m0 + j + half * 8;
    const float r = sigmoid_f(acc_r[j] + br);
    const float z = sigmoid_f(acc_z[j] + bz);
    const float n = tanhf(acc_in[j] + bi + r * (acc_hn[j] + bh));
    const float hp = hprev[(size_t)m * HID + col];
    hout[(size_t)m * HID + col] = (1.0f - z) * n + z * hp;
  }
}

// ---------------------------------------------------------------------------
extern "C" void kernel_launch(void* const* d_in, const int* in_sizes, int n_in,
                              void* d_out, int out_size, void* d_ws, size_t ws_size,
                              hipStream_t stream) {
  (void)in_sizes; (void)n_in; (void)out_size;

  const float* c_inp    = (const float*)d_in[0];
  const float* inp      = (const float*)d_in[1];
  const float* gru_init = (const float*)d_in[2];
  const float* att_init = (const float*)d_in[3];
  const float* Wa = (const float*)d_in[4];  const float* ba = (const float*)d_in[5];
  const float* Wb = (const float*)d_in[6];  const float* bb = (const float*)d_in[7];
  const float* Wk = (const float*)d_in[8];  const float* bk = (const float*)d_in[9];
  const float* Wih = (const float*)d_in[10]; const float* bih = (const float*)d_in[11];
  const float* Whh = (const float*)d_in[12]; const float* bhh = (const float*)d_in[13];

  float* out  = (float*)d_out;
  float* hid  = out;                                    // [TS,B,H]
  float* attk = out + (size_t)TSN * BATCH * HID;        // [TS,B,K]
  float* attw = attk + (size_t)TSN * BATCH * KMIX;      // [TS,B,H]

  // Workspace layout (bf16): swizzled weights | w_bf | h_bf[0] | h_bf[1]
  bf16_t* swW  = (bf16_t*)d_ws;
  bf16_t* w_bf = swW + WFRAG_ELEMS;
  bf16_t* h_bf = w_bf + (size_t)BATCH * HID;
  const size_t need = (WFRAG_ELEMS + (size_t)3 * BATCH * HID) * sizeof(bf16_t);
  const bool fast = ws_size >= need;

  if (fast) {
    prep_weights<<<384, 256, 0, stream>>>(Wih, Whh, swW);
    prep_h0<<<(BATCH * HID) / 256, 256, 0, stream>>>(gru_init, h_bf);
  }

  for (int t = 0; t < TSN; ++t) {
    const float* x     = inp + (size_t)t * BATCH * HID;
    const float* kprev = (t == 0) ? att_init : attk + (size_t)(t - 1) * BATCH * KMIX;
    float* kout = attk + (size_t)t * BATCH * KMIX;
    float* wout = attw + (size_t)t * BATCH * HID;

    attn_step<<<BATCH, 256, 0, stream>>>(x, Wa, ba, Wb, bb, Wk, bk,
                                         kprev, c_inp, kout, wout,
                                         fast ? w_bf : (bf16_t*)nullptr);

    const float* hprev = (t == 0) ? gru_init : hid + (size_t)(t - 1) * BATCH * HID;
    float* hcur = hid + (size_t)t * BATCH * HID;

    if (fast) {
      bf16_t* h_in  = h_bf + (size_t)(t % 2) * BATCH * HID;
      bf16_t* h_out = h_bf + (size_t)((t + 1) % 2) * BATCH * HID;
      gru_step_bf<<<dim3(BATCH / 16, HID / 16), 32, 0, stream>>>(
          w_bf, h_in, hprev, swW, bih, bhh, hcur, h_out);
    } else {
      gru_step_f32<<<dim3(BATCH / 16, HID / 16), 32, 0, stream>>>(
          wout, hprev, Wih, bih, Whh, bhh, hcur);
    }
  }
}
